// LinearScaleWithWeights_23424751632773
// MI455X (gfx1250) — matrix operand order repaired and checked by reading, not compiled, not run on hardware
//
#include <hip/hip_runtime.h>

// ---------------- CDNA5 (gfx1250) types ----------------
typedef __attribute__((ext_vector_type(16))) __bf16 v16bf;
typedef __attribute__((ext_vector_type(8)))  __bf16 v8bf;
typedef __attribute__((ext_vector_type(8)))  float  v8f;

union FragAB { v16bf v; v8bf h[2]; };

#define DIM_TOT 1184
#define WNUM_TOT 480
#define NT 256
#define NWAVES 8

// One irrep block:  out[n, XO + w*D + i] = (1/M) * sum_u x[n, XO + u*D + i] *
//                                           wgt[n, WO + u] * W[u*M + w]
// GEMM view: A[(n,i), u] (scaled x) x B[u, w] = W.  K = M, cols = M,
// rows = S*D per workgroup.  bf16 WMMA 16x16x32, f32 accumulate.
// Each wave iteration computes a 16x32 output tile (two independent
// accumulator chains sharing one A fragment -> 3 ds_load_b128 per wmma).
template <int M, int D, int XO, int WO, int S, int RT>
__global__ __launch_bounds__(NT) void tp_linear_wmma_kernel(
    const float* __restrict__ x, const float* __restrict__ wgt,
    const float* __restrict__ Wm, float* __restrict__ out, int n_total) {
  constexpr int CT = M / 16;            // 16-wide column tiles (even for all blocks)
  constexpr int NP = CT / 2;            // column-tile pairs per row tile
  constexpr int STRIDE = M + 8;         // padded LDS row stride (bf16 elems)
  constexpr int ROWS = S * D;           // valid GEMM rows per WG
  constexpr float inv_m = 1.0f / (float)M;

  extern __shared__ __align__(16) char smem_raw[];
  __bf16* Wt   = (__bf16*)smem_raw;             // [M][STRIDE]  Wt[w][u] = W[u][w]
  __bf16* Alds = Wt + (size_t)M * STRIDE;       // [RT*16][STRIDE]

  const int tid = threadIdx.x;
  const int n0 = blockIdx.x * S;

  // ---- Phase 1a: stage W transposed as bf16 (coalesced global reads) ----
  for (int p = tid; p < M * M; p += NT) {
    int u = p / M, w = p - u * M;
    Wt[w * STRIDE + u] = (__bf16)Wm[p];
  }
  // ---- Phase 1b: stage scaled activations A[(s*D+i)][u] = x*w/M ----
  for (int p = tid; p < S * M * D; p += NT) {
    int s = p / (M * D);
    int rem = p - s * (M * D);          // = u*D + i, contiguous in x block
    int u = rem / D, i = rem - u * D;
    int n = n0 + s;
    float v = 0.0f;
    if (n < n_total)
      v = x[(size_t)n * DIM_TOT + XO + rem] *
          wgt[(size_t)n * WNUM_TOT + WO + u] * inv_m;
    Alds[(s * D + i) * STRIDE + u] = (__bf16)v;
  }
  // ---- Phase 1c: zero the padded rows so WMMA tiles read zeros ----
  for (int p = tid; p < (RT * 16 - ROWS) * M; p += NT) {
    int r = ROWS + p / M, u = p - (p / M) * M;
    Alds[r * STRIDE + u] = (__bf16)0.0f;
  }
  __syncthreads();

  // ---- Phase 2: WMMA GEMM, one 16x32 tile per wave iteration ----
  const int wave = tid >> 5;
  const int lane = tid & 31;
  const int lrow = lane & 15;
  const int half = lane >> 4;           // 0: lanes 0-15, 1: lanes 16-31
  const int kb_a = half * 8;            // A 16x32 bf16 lane map (ISA 7.12.2)
  const int kb_b = half * 8;            // B 32x16 bf16 lane map

  for (int t = wave; t < RT * NP; t += NWAVES) {
    const int rt = t / NP, cp = t - rt * NP;
    const __bf16* arow  = Alds + (rt * 16 + lrow) * STRIDE;
    const __bf16* brow0 = Wt + (cp * 32 + lrow) * STRIDE;
    const __bf16* brow1 = brow0 + 16 * STRIDE;

    v8f acc0 = {}, acc1 = {};
#pragma unroll
    for (int k0 = 0; k0 < M; k0 += 32) {
      FragAB a, b0, b1;
      a.h[0]  = *(const v8bf*)(arow + k0 + kb_a);        // K = k0+kb_a .. +7
      a.h[1]  = *(const v8bf*)(arow + k0 + 16 + kb_a);   // K = k0+16+kb_a .. +7
      b0.h[0] = *(const v8bf*)(brow0 + k0 + kb_b);       // K = k0+kb_b .. +7
      b0.h[1] = *(const v8bf*)(brow0 + k0 + kb_b + 16);  // K = k0+16+kb_b .. +7
      b1.h[0] = *(const v8bf*)(brow1 + k0 + kb_b);
      b1.h[1] = *(const v8bf*)(brow1 + k0 + kb_b + 16);
      acc0 = __builtin_amdgcn_wmma_f32_16x16x32_bf16(
          false, a.v, false, b0.v, (short)0, acc0, false, false);
      acc1 = __builtin_amdgcn_wmma_f32_16x16x32_bf16(
          false, a.v, false, b1.v, (short)0, acc1, false, false);
    }

    // D: VGPR r holds row (r + 8*half), column = lrow  (ISA 7.12.2)
    const int col0 = cp * 32 + lrow;    // output channel w (first tile)
    const int Rbase = rt * 16 + half * 8;
    // Uniform fast path: entire 16-row tile maps to valid samples.
    const bool full = ((rt + 1) * 16 <= ROWS) &&
                      (n0 + (rt * 16 + 15) / D < n_total);
    if (full) {
#pragma unroll
      for (int r = 0; r < 8; ++r) {
        int R = Rbase + r;
        int s = R / D, i = R - s * D;
        float* o = out + (size_t)(n0 + s) * DIM_TOT + XO + i;
        o[col0 * D] = acc0[r];
        o[(col0 + 16) * D] = acc1[r];
      }
    } else {
#pragma unroll
      for (int r = 0; r < 8; ++r) {
        int R = Rbase + r;
        if (R < ROWS) {
          int s = R / D, i = R - s * D;
          int n = n0 + s;
          if (n < n_total) {
            float* o = out + (size_t)n * DIM_TOT + XO + i;
            o[col0 * D] = acc0[r];
            o[(col0 + 16) * D] = acc1[r];
          }
        }
      }
    }
  }
}

extern "C" void kernel_launch(void* const* d_in, const int* in_sizes, int n_in,
                              void* d_out, int out_size, void* d_ws, size_t ws_size,
                              hipStream_t stream) {
  (void)n_in; (void)out_size; (void)d_ws; (void)ws_size;
  const float* x   = (const float*)d_in[0];
  const float* wgt = (const float*)d_in[1];
  const float* W0  = (const float*)d_in[2];
  const float* W1  = (const float*)d_in[3];
  const float* W2  = (const float*)d_in[4];
  const float* W3  = (const float*)d_in[5];
  float* out = (float*)d_out;
  const int n = in_sizes[0] / DIM_TOT;

  // Block configs: (M, D, XO, WO, S samples/WG, RT row-tiles)
  // b0: 256x0e  b1: 128x1o  b2: 64x2e  b3: 32x3o
  auto k0 = tp_linear_wmma_kernel<256, 1,   0,   0, 64, 4>;
  auto k1 = tp_linear_wmma_kernel<128, 3, 256, 256, 21, 4>;
  auto k2 = tp_linear_wmma_kernel< 64, 5, 640, 384, 13, 5>;
  auto k3 = tp_linear_wmma_kernel< 32, 7, 960, 448,  9, 4>;

  const int lds0 = (256 * (256 + 8) + 4 * 16 * (256 + 8)) * 2; // 168960 B
  const int lds1 = (128 * (128 + 8) + 4 * 16 * (128 + 8)) * 2; //  52224 B
  const int lds2 = ( 64 * ( 64 + 8) + 5 * 16 * ( 64 + 8)) * 2; //  20736 B
  const int lds3 = ( 32 * ( 32 + 8) + 4 * 16 * ( 32 + 8)) * 2; //   7680 B

  (void)hipFuncSetAttribute((const void*)k0, hipFuncAttributeMaxDynamicSharedMemorySize, lds0);
  (void)hipFuncSetAttribute((const void*)k1, hipFuncAttributeMaxDynamicSharedMemorySize, lds1);
  (void)hipFuncSetAttribute((const void*)k2, hipFuncAttributeMaxDynamicSharedMemorySize, lds2);
  (void)hipFuncSetAttribute((const void*)k3, hipFuncAttributeMaxDynamicSharedMemorySize, lds3);

  k0<<<(n + 63) / 64, NT, lds0, stream>>>(x, wgt, W0, out, n);
  k1<<<(n + 20) / 21, NT, lds1, stream>>>(x, wgt, W1, out, n);
  k2<<<(n + 12) / 13, NT, lds2, stream>>>(x, wgt, W2, out, n);
  k3<<<(n +  8) /  9, NT, lds3, stream>>>(x, wgt, W3, out, n);
}
